// QConv2d_70832600646421
// MI455X (gfx1250) — matrix operand order repaired and checked by reading, not compiled, not run on hardware
//
#include <hip/hip_runtime.h>

typedef __attribute__((ext_vector_type(16))) _Float16 v16h;
typedef __attribute__((ext_vector_type(8)))  float    v8f;
typedef __attribute__((ext_vector_type(4)))  int      v4i;

#define N_IMG 8
#define H_IMG 64
#define W_IMG 64
#define C_IMG 16
#define NCLUS 32
#define D_REAL 145            // 3*3*16 + 1
#define D_PAD  160            // padded to multiple of 32 (WMMA K)
#define P_TOT  (N_IMG * H_IMG * W_IMG)   // 32768 pixels
#define TILE_P 128            // pixels per block (8 waves x 16 rows)
#define LDS_STRIDE 168        // halves, padded row stride (336 B, conflict-friendly)
#define NT (D_PAD / 16)       // 10 column tiles
#define KT (D_PAD / 32)       // 5 k tiles

#define PE_TILE_BYTES (TILE_P * LDS_STRIDE * 2)   // 43008
#define QBUF_BYTES    (D_PAD  * LDS_STRIDE * 2)   // 53760

union Frag16 { v16h h; uint4 u[2]; };   // 32 bytes

// ---- CDNA5 async global->LDS path (ASYNCcnt), with safe fallback ----------
#if defined(__gfx1250__) && __has_builtin(__builtin_amdgcn_global_load_async_to_lds_b128)
#define HAVE_ASYNC 1
#else
#define HAVE_ASYNC 0
#endif

#if HAVE_ASYNC
// Param types per hipcc diagnostic: (int4 [AS1]*, int4 [AS3]*, imm int, imm int)
#define ASYNC_GPTR(p) ((__attribute__((address_space(1))) v4i*)(unsigned long long)(p))
#define ASYNC_LPTR(p) ((__attribute__((address_space(3))) v4i*)(unsigned int)(unsigned long long)(p))
#if __has_builtin(__builtin_amdgcn_s_wait_asynccnt)
#define WAIT_ASYNC() __builtin_amdgcn_s_wait_asynccnt(0)
#else
#define WAIT_ASYNC() asm volatile("s_wait_asynccnt 0x0" ::: "memory")
#endif
#else
#define WAIT_ASYNC() ((void)0)
#endif

// ---------------------------------------------------------------------------
// Prep 1: extract 3x3 SAME patches + bias slot, cast f32 -> f16, zero-pad to 160
// PE[p][d], row-major, contiguous.
// ---------------------------------------------------------------------------
__global__ void build_pe(const float* __restrict__ x, _Float16* __restrict__ pe) {
    int g = blockIdx.x * blockDim.x + threadIdx.x;   // grid sized exactly
    int p = g / D_PAD;
    int d = g - p * D_PAD;
    int n = p >> 12;            // / (64*64)
    int h = (p >> 6) & 63;
    int w = p & 63;
    float v = 0.0f;
    if (d < 144) {              // (kh, kw, C) flattened, TF order
        int kh = d / 48;
        int kw = (d / 16) % 3;
        int cc = d & 15;
        int hh = h + kh - 1, ww = w + kw - 1;
        if (hh >= 0 && hh < H_IMG && ww >= 0 && ww < W_IMG)
            v = x[(((n * H_IMG + hh) * W_IMG + ww) * C_IMG) + cc];
    } else if (d == 144) {
        v = 1.0f;               // bias-1 slot
    }
    pe[g] = (_Float16)v;
}

// ---------------------------------------------------------------------------
// Prep 2: quadB[c][n][k] = quad[c][k][n]  (transposed, f16, zero-padded 145->160)
// Layout matches WMMA B-fragment loads: for column n, k runs contiguous.
// ---------------------------------------------------------------------------
__global__ void build_quadB(const float* __restrict__ quad, _Float16* __restrict__ qB) {
    int g = blockIdx.x * blockDim.x + threadIdx.x;   // grid sized exactly
    int c = g / (D_PAD * D_PAD);
    int rem = g - c * D_PAD * D_PAD;
    int nn = rem / D_PAD;        // output column e
    int kk = rem - nn * D_PAD;   // inner index d
    float v = 0.0f;
    if (nn < D_REAL && kk < D_REAL)
        v = quad[(c * D_REAL + kk) * D_REAL + nn];
    qB[g] = (_Float16)v;
}

// ---------------------------------------------------------------------------
// Stage quad[c] (160 rows x 20 uint4 = 51.2 KB) into an LDS buffer.
// Async (ASYNCcnt-tracked) when available; consumer does WAIT_ASYNC + barrier.
// ---------------------------------------------------------------------------
__device__ __forceinline__ void stage_quad(const _Float16* __restrict__ qB, int c,
                                           char* dstBase, int tid) {
    const uint4* qsrc = (const uint4*)(qB + (size_t)c * (D_PAD * D_PAD));
    #pragma unroll
    for (int i = 0; i < 13; ++i) {
        int idx = tid + i * 256;
        if (idx < 3200) {
            int r  = idx / 20;
            int c8 = idx - r * 20;
            char* dst = dstBase + r * (LDS_STRIDE * 2) + c8 * 16;
#if HAVE_ASYNC
            __builtin_amdgcn_global_load_async_to_lds_b128(
                ASYNC_GPTR(qsrc + idx), ASYNC_LPTR(dst), 0, 0);
#else
            *(uint4*)dst = qsrc[idx];
#endif
        }
    }
}

// ---------------------------------------------------------------------------
// Main: per block of 128 pixels, loop clusters; Y = PE_tile x quad[c] via WMMA,
// q[p,c] = -sum_n Y[p,n] * pe[p,n]  (f32 accumulate + 16-lane shfl reduce).
// quad[c+1] streams into the alternate LDS buffer while cluster c computes.
// ---------------------------------------------------------------------------
__global__ __launch_bounds__(256, 1) void qconv_wmma(
        const _Float16* __restrict__ pe,
        const _Float16* __restrict__ qB,
        float* __restrict__ out) {
    extern __shared__ char smem[];
    _Float16* sPE = (_Float16*)smem;                    // 128 x 168 halves
    char* sQ0 = smem + PE_TILE_BYTES;                   // 160 x 168 halves
    char* sQ1 = sQ0 + QBUF_BYTES;                       // 160 x 168 halves

    const int tid  = threadIdx.x;
    const int lane = tid & 31;
    const int wave = tid >> 5;
    const int pb   = blockIdx.x * TILE_P;

    // ---- stage PE tile: 128*160 halves = 40960 B contiguous = 2560 uint4 ----
    {
        const uint4* src = (const uint4*)(pe + (size_t)pb * D_PAD);  // 20 uint4/row
        #pragma unroll
        for (int i = 0; i < 10; ++i) {
            int idx = tid + i * 256;
            int r   = idx / 20;
            int c8  = idx - r * 20;
            uint4 v = src[idx];
            *(uint4*)((char*)sPE + r * (LDS_STRIDE * 2) + c8 * 16) = v;
        }
    }
    // prologue: start streaming cluster 0's quad matrix
    stage_quad(qB, 0, sQ0, tid);
    __syncthreads();

    const int rb      = wave * 16;     // this wave's pixel-row base within tile
    const int halfSel = lane >> 4;     // 0: lanes 0-15, 1: lanes 16-31
    const int mlo     = lane & 15;

    // ---- hoist A fragments (16 rows x 32 k) for all 5 k-tiles ----
    // ISA A layout: lane holds row = lane%16; elems 0..7 -> k0..k0+7,
    // elems 8..15 -> k0+16..k0+23, k0 = 32*kt + 8*(lane/16)
    Frag16 A[KT];
    #pragma unroll
    for (int kt = 0; kt < KT; ++kt) {
        const char* base = (const char*)sPE + (rb + mlo) * (LDS_STRIDE * 2)
                           + (kt * 32 + halfSel * 8) * 2;
        A[kt].u[0] = *(const uint4*)(base);
        A[kt].u[1] = *(const uint4*)(base + 32);
    }

    // ---- hoist pe factors in C-matrix layout: pf[nt][i] = pe[row(i), col(nt)] ----
    float pf[NT][8];
    #pragma unroll
    for (int nt = 0; nt < NT; ++nt) {
        #pragma unroll
        for (int i = 0; i < 8; ++i) {
            int row = rb + i + 8 * halfSel;
            int col = nt * 16 + mlo;
            pf[nt][i] = (float)sPE[row * LDS_STRIDE + col];
        }
    }

    for (int c = 0; c < NCLUS; ++c) {
        char* cur = (c & 1) ? sQ1 : sQ0;
        char* nxt = (c & 1) ? sQ0 : sQ1;
        WAIT_ASYNC();        // my copies for cluster c complete
        __syncthreads();     // everyone's copies landed; iter c-1 reads of nxt done
        if (c + 1 < NCLUS)
            stage_quad(qB, c + 1, nxt, tid);   // overlaps with compute below

        float R[8] = {0.f, 0.f, 0.f, 0.f, 0.f, 0.f, 0.f, 0.f};
        #pragma unroll
        for (int nt = 0; nt < NT; ++nt) {
            v8f acc = {};
            #pragma unroll
            for (int kt = 0; kt < KT; ++kt) {
                // B layout: lane holds col n = lane%16; elems 0..7 -> k0..k0+7,
                // elems 8..15 -> k0+16..k0+23, k0 = 32*kt + 8*(lane/16)
                Frag16 B;
                const char* bb = cur + (nt * 16 + mlo) * (LDS_STRIDE * 2)
                                 + (kt * 32 + halfSel * 8) * 2;
                B.u[0] = *(const uint4*)(bb);
                B.u[1] = *(const uint4*)(bb + 32);
                acc = __builtin_amdgcn_wmma_f32_16x16x32_f16(
                    false, A[kt].h, false, B.h, (short)0, acc, false, false);
            }
            #pragma unroll
            for (int i = 0; i < 8; ++i)
                R[i] += acc[i] * pf[nt][i];     // fold pe[p,n] factor, f32
        }

        // ---- per-row sum: reduce over the 16 lanes of each half-wave ----
        #pragma unroll
        for (int i = 0; i < 8; ++i) {
            float r = R[i];
            r += __shfl_xor(r, 1, 32);
            r += __shfl_xor(r, 2, 32);
            r += __shfl_xor(r, 4, 32);
            r += __shfl_xor(r, 8, 32);
            if (mlo == i) {   // lanes i and i+16 write rows rb+i and rb+i+8
                int row = pb + rb + i + 8 * halfSel;
                out[row * NCLUS + c] = -r;       // is_neg = 1
            }
        }
    }
}

extern "C" void kernel_launch(void* const* d_in, const int* in_sizes, int n_in,
                              void* d_out, int out_size, void* d_ws, size_t ws_size,
                              hipStream_t stream) {
    (void)in_sizes; (void)n_in; (void)out_size; (void)ws_size;
    const float* x    = (const float*)d_in[0];   // (8,64,64,16) f32
    const float* quad = (const float*)d_in[1];   // (32,145,145) f32
    float* out = (float*)d_out;                  // (8,64,64,32) f32

    _Float16* pe = (_Float16*)d_ws;                          // 32768*160 halves
    _Float16* qB = pe + (size_t)P_TOT * D_PAD;               // 32*160*160 halves

    build_pe<<<(P_TOT * D_PAD) / 256, 256, 0, stream>>>(x, pe);
    build_quadB<<<(NCLUS * D_PAD * D_PAD) / 256, 256, 0, stream>>>(quad, qB);

    size_t shmem = (size_t)PE_TILE_BYTES + 2 * (size_t)QBUF_BYTES;  // 150528 B
    (void)hipFuncSetAttribute(reinterpret_cast<const void*>(qconv_wmma),
                              hipFuncAttributeMaxDynamicSharedMemorySize,
                              (int)shmem);
    qconv_wmma<<<P_TOT / TILE_P, 256, shmem, stream>>>(pe, qB, out);
}
